// CCepLTVFilter_56384330662031
// MI455X (gfx1250) — matrix operand order repaired and checked by reading, not compiled, not run on hardware
//
#include <hip/hip_runtime.h>

typedef __attribute__((ext_vector_type(16))) _Float16 v16h;
typedef __attribute__((ext_vector_type(8)))  _Float16 v8h;
typedef __attribute__((ext_vector_type(8)))  float    v8f;

#define T_FRAMES 256
#define FFTN     1024
#define ZLEN     65536

// ---------------------------------------------------------------------------
// Pack conv weights (O,I,3) f32 -> B-operand layout [N_pad][Kc_pad] f16,
// with c = k*I + i (must match the im2col packing of A).
// ---------------------------------------------------------------------------
__global__ void pack_w_f16(const float* __restrict__ w, _Float16* __restrict__ Bp,
                           int O, int I, int Kcp, int total)
{
  int idx = blockIdx.x * blockDim.x + threadIdx.x;
  if (idx >= total) return;
  int o = idx / Kcp;
  int c = idx - o * Kcp;
  float v = 0.f;
  int CK = I * 3;
  if (o < O && c < CK) {
    int k = c / I;
    int i = c - k * I;
    v = w[(o * I + i) * 3 + k];
  }
  Bp[idx] = (_Float16)v;
}

// ---------------------------------------------------------------------------
// Pack activations (T x CIN f32) -> im2col A matrix [T][Kc_pad] f16,
// c = k*CIN + i, 'same' padding for K=3 (row t reads src rows t-1..t+1).
// ---------------------------------------------------------------------------
template<int CIN>
__global__ void pack_a_f16(const float* __restrict__ src, _Float16* __restrict__ Ap,
                           int Kcp, int total)
{
  int idx = blockIdx.x * blockDim.x + threadIdx.x;
  if (idx >= total) return;
  int t = idx / Kcp;
  int c = idx - t * Kcp;
  float v = 0.f;
  if (c < CIN * 3) {
    int kw = c / CIN;                 // compile-time CIN -> shifts/magic once
    int ci = c - kw * CIN;
    int tt = t + kw - 1;
    if (tt >= 0 && tt < T_FRAMES) v = src[tt * CIN + ci];
  }
  Ap[idx] = (_Float16)v;
}

// ---------------------------------------------------------------------------
// GEMM via WMMA: one wave computes a 32x16 tile (two 16x16 M-tiles sharing
// one B fragment). A and B are pre-packed f16, K-major rows.
// A-frag (16-bit A 16x32): lane {m=lane&15, kh=lane>>4}; two contiguous runs:
//   a[0..7]  -> K = k0 + kh*8 + j
//   a[8..15] -> K = k0 + 16 + kh*8 + (j-8)
// B-frag (16-bit B 32x16): lane {n=lane&15, kh}; same K runs as A-frag.
// C/D: lane {n=lane&15, mh=lane>>4}; c[v] -> row mh*8+v.
// ---------------------------------------------------------------------------
template<bool RELU, bool QDIV>
__global__ __launch_bounds__(128)
void gemm_wmma_f16(const _Float16* __restrict__ Ap,   // 256 x Kcp
                   const _Float16* __restrict__ Bp,   // Np x Kcp
                   const float* __restrict__ bias,    // Ocnt
                   const float* __restrict__ qn,      // Ocnt divisor (or null)
                   float* __restrict__ dst,           // 256 x Ostride
                   int Kcp, int nTilesN, int Ocnt, int Ostride)
{
  const int wid   = blockIdx.x * 4 + (threadIdx.x >> 5);
  const int lane  = threadIdx.x & 31;
  const int tM2   = wid / nTilesN;          // 32-row tile index
  const int tN    = wid - tM2 * nTilesN;
  const int lan15 = lane & 15;
  const int kh    = lane >> 4;
  const int t0    = tM2 << 5;
  const int n0    = tN << 4;
  const int brow  = n0 + lan15;

  const _Float16* aptr0 = Ap + (size_t)(t0 + lan15) * Kcp + kh * 8;
  const _Float16* aptr1 = aptr0 + (size_t)16 * Kcp;
  const _Float16* bptr  = Bp + (size_t)brow * Kcp + kh * 8;

  v8f acc0 = {}, acc1 = {};
  for (int k0 = 0; k0 < Kcp; k0 += 32) {
    if (k0 + 32 < Kcp) __builtin_prefetch(bptr + k0 + 32, 0, 3);
    v8h a00 = *(const v8h*)(aptr0 + k0);
    v8h a01 = *(const v8h*)(aptr0 + k0 + 16);
    v8h a10 = *(const v8h*)(aptr1 + k0);
    v8h a11 = *(const v8h*)(aptr1 + k0 + 16);
    v8h b0 = *(const v8h*)(bptr + k0);
    v8h b1 = *(const v8h*)(bptr + k0 + 16);
    v16h b  = __builtin_shufflevector(b0, b1, 0,1,2,3,4,5,6,7,8,9,10,11,12,13,14,15);
    v16h a0 = __builtin_shufflevector(a00, a01, 0,1,2,3,4,5,6,7,8,9,10,11,12,13,14,15);
    v16h a1 = __builtin_shufflevector(a10, a11, 0,1,2,3,4,5,6,7,8,9,10,11,12,13,14,15);
    acc0 = __builtin_amdgcn_wmma_f32_16x16x32_f16(false, a0, false, b, (short)0, acc0, false, false);
    acc1 = __builtin_amdgcn_wmma_f32_16x16x32_f16(false, a1, false, b, (short)0, acc1, false, false);
  }

  if (brow < Ocnt) {
    float bv = bias[brow];
    float qv = QDIV ? qn[brow] : 1.0f;
#pragma unroll
    for (int v = 0; v < 8; ++v) {
      int r0 = t0 + kh * 8 + v;
      float v0 = acc0[v] + bv;
      float v1 = acc1[v] + bv;
      if (RELU) { v0 = fmaxf(v0, 0.f); v1 = fmaxf(v1, 0.f); }
      if (QDIV) { v0 /= qv; v1 /= qv; }
      dst[r0 * Ostride + brow] = v0;
      dst[(r0 + 16) * Ostride + brow] = v1;
    }
  }
}

// ---------------------------------------------------------------------------
// In-LDS radix-2 DIT 1024-pt FFT (input already bit-reversed). 256 threads,
// 2 butterflies/thread/stage. Forward twiddle e^{-i*2pi*pos/len}.
// ---------------------------------------------------------------------------
__device__ __forceinline__ void fft1024(float* re, float* im, int tid)
{
#pragma unroll 1
  for (int s = 1; s <= 10; ++s) {
    const int half = 1 << (s - 1);
#pragma unroll
    for (int r = 0; r < 2; ++r) {
      int b   = tid + (r << 8);
      int grp = b >> (s - 1);
      int pos = b & (half - 1);
      int i0  = (grp << s) + pos;
      int i1  = i0 + half;
      float ang = -3.14159265358979323846f * (float)pos / (float)half;
      float wr = __cosf(ang);
      float wi = __sinf(ang);
      float xr = re[i1], xi = im[i1];
      float tr = wr * xr - wi * xi;
      float ti = wr * xi + wi * xr;
      float ur = re[i0], ui = im[i0];
      re[i0] = ur + tr; im[i0] = ui + ti;
      re[i1] = ur - tr; im[i1] = ui - ti;
    }
    __syncthreads();
  }
}

// ccep (256 x stride224, 222 valid) -> imp (256 x 1024)
//   cc = pad(ccep, 401 each side); Y = fft(cc);
//   g = 10^(Y.re/10) * e^{i*Y.im}; imp = Re(ifft(g)) = Re(fft(conj g))/1024
__global__ __launch_bounds__(256)
void cepstrum_fft(const float* __restrict__ ccep, float* __restrict__ imp)
{
  __shared__ float re[FFTN];
  __shared__ float im[FFTN];
  const int t   = blockIdx.x;
  const int tid = threadIdx.x;
#pragma unroll
  for (int q = 0; q < 4; ++q) {
    int j  = tid + (q << 8);
    int br = __brev((unsigned)j) >> 22;
    float v = 0.f;
    if (j >= 401 && j < 623) v = ccep[t * 224 + (j - 401)];
    re[br] = v;
    im[br] = 0.f;
  }
  __syncthreads();
  fft1024(re, im, tid);

  float gr[4], gi[4];
#pragma unroll
  for (int q = 0; q < 4; ++q) {
    int j = tid + (q << 8);
    float mag = __expf(re[j] * 0.23025850929940457f);  // 10^(x/10)
    float ph  = im[j];
    float sn, cs;
    __sincosf(ph, &sn, &cs);
    gr[q] = mag * cs;          // conj(g)
    gi[q] = -mag * sn;
  }
  __syncthreads();
#pragma unroll
  for (int q = 0; q < 4; ++q) {
    int j  = tid + (q << 8);
    int br = __brev((unsigned)j) >> 22;
    re[br] = gr[q];
    im[br] = gi[q];
  }
  __syncthreads();
  fft1024(re, im, tid);
#pragma unroll
  for (int q = 0; q < 4; ++q) {
    int j = tid + (q << 8);
    imp[t * FFTN + j] = re[j] * (1.0f / 1024.0f);
  }
}

// ---------------------------------------------------------------------------
// LTV filter (batched Toeplitz matvec per frame) + Hann window.
//   out[t,w] = win[w] * sum_u frame[t,u] * imp[t, 511+u-w]
//   frame[t,u] = z[t*256 + u - 255] (zero-padded)
// ---------------------------------------------------------------------------
__global__ __launch_bounds__(256)
void ltv_filter(const float* __restrict__ imp, const float* __restrict__ z,
                const float* __restrict__ win, float* __restrict__ outw)
{
  __shared__ float sImp[FFTN];
  __shared__ float sFrm[512];
  const int t   = blockIdx.x;
  const int tid = threadIdx.x;
#pragma unroll
  for (int q = 0; q < 4; ++q) sImp[tid + (q << 8)] = imp[t * FFTN + tid + (q << 8)];
#pragma unroll
  for (int q = 0; q < 2; ++q) {
    int u  = tid + (q << 8);
    int zi = t * 256 + u - 255;
    sFrm[u] = (zi >= 0 && zi < ZLEN) ? z[zi] : 0.f;
  }
  __syncthreads();
#pragma unroll
  for (int h = 0; h < 2; ++h) {
    int w = tid + (h << 8);
    float acc = 0.f;
    const float* ip = sImp + 511 - w;
#pragma unroll 4
    for (int u = 0; u < 512; ++u) acc = fmaf(sFrm[u], ip[u], acc);
    outw[t * 512 + w] = acc * win[w];
  }
}

// Overlap-add with jnp.roll semantics: s[t*256+w] = out[t,w] + out[(t-1)%256, 256+w]
__global__ __launch_bounds__(256)
void overlap_add(const float* __restrict__ outw, float* __restrict__ s)
{
  int n  = blockIdx.x * blockDim.x + threadIdx.x;
  int t  = n >> 8;
  int w  = n & 255;
  int tp = (t + 255) & 255;
  s[n] = outw[t * 512 + w] + outw[tp * 512 + 256 + w];
}

// 128-tap post-filter, 'same' padding (63 left, 64 right):
//   y[n] = pf_b + sum_k s[n-63+k] * pf_w[k]
__global__ __launch_bounds__(256)
void post_filter(const float* __restrict__ s, const float* __restrict__ pfw,
                 const float* __restrict__ pfb, float* __restrict__ y)
{
  __shared__ float tile[256 + 127];
  const int base = blockIdx.x * 256;
  const int tid  = threadIdx.x;
  for (int q = tid; q < 256 + 127; q += 256) {
    int idx = base - 63 + q;
    tile[q] = (idx >= 0 && idx < ZLEN) ? s[idx] : 0.f;
  }
  __syncthreads();
  float acc = pfb[0];
#pragma unroll 8
  for (int k = 0; k < 128; ++k) acc = fmaf(tile[tid + k], pfw[k], acc);
  y[base + tid] = acc;
}

// ---------------------------------------------------------------------------
extern "C" void kernel_launch(void* const* d_in, const int* in_sizes, int n_in,
                              void* d_out, int out_size, void* d_ws, size_t ws_size,
                              hipStream_t stream)
{
  (void)in_sizes; (void)n_in; (void)out_size; (void)ws_size;
  const float* x    = (const float*)d_in[0];   // (1,256,80)
  const float* z    = (const float*)d_in[1];   // (1,1,65536)
  const float* w1   = (const float*)d_in[2];   // (256,80,3)
  const float* b1   = (const float*)d_in[3];
  const float* w2   = (const float*)d_in[4];   // (256,256,3)
  const float* b2   = (const float*)d_in[5];
  const float* w3   = (const float*)d_in[6];   // (222,256,3)
  const float* b3   = (const float*)d_in[7];
  const float* pfw  = (const float*)d_in[8];   // (1,1,128)
  const float* pfb  = (const float*)d_in[9];
  const float* quef = (const float*)d_in[10];  // (222,)
  const float* win  = (const float*)d_in[11];  // (512,)
  float* yout = (float*)d_out;                 // 65536 f32

  char* ws = (char*)d_ws;
  const size_t OFF_B1   = 0;                         // 256*256 f16
  const size_t OFF_B2   = OFF_B1 + 256u*256*2;       // 256*768 f16
  const size_t OFF_B3   = OFF_B2 + 256u*768*2;       // 224*768 f16
  const size_t OFF_A1   = OFF_B3 + 224u*768*2;       // 256*256 f16
  const size_t OFF_A2   = OFF_A1 + 256u*256*2;       // 256*768 f16
  const size_t OFF_A3   = OFF_A2 + 256u*768*2;       // 256*768 f16
  const size_t OFF_H1   = OFF_A3 + 256u*768*2;       // 256*256 f32
  const size_t OFF_H2   = OFF_H1 + 256u*256*4;       // 256*256 f32
  const size_t OFF_CC   = OFF_H2 + 256u*256*4;       // 256*224 f32 (222 valid)
  const size_t OFF_IMP  = OFF_CC + 256u*224*4;       // 256*1024 f32
  const size_t OFF_OUTW = OFF_IMP + 256u*1024*4;     // 256*512 f32
  const size_t OFF_S    = OFF_OUTW + 256u*512*4;     // 65536 f32

  _Float16* B1p = (_Float16*)(ws + OFF_B1);
  _Float16* B2p = (_Float16*)(ws + OFF_B2);
  _Float16* B3p = (_Float16*)(ws + OFF_B3);
  _Float16* A1p = (_Float16*)(ws + OFF_A1);
  _Float16* A2p = (_Float16*)(ws + OFF_A2);
  _Float16* A3p = (_Float16*)(ws + OFF_A3);
  float* h1   = (float*)(ws + OFF_H1);
  float* h2   = (float*)(ws + OFF_H2);
  float* cc   = (float*)(ws + OFF_CC);
  float* imp  = (float*)(ws + OFF_IMP);
  float* outw = (float*)(ws + OFF_OUTW);
  float* sbuf = (float*)(ws + OFF_S);

  // Pack weights into f16 B operands (tiny, fully parallel).
  pack_w_f16<<<(256*256 + 255)/256, 256, 0, stream>>>(w1, B1p, 256,  80, 256, 256*256);
  pack_w_f16<<<(256*768 + 255)/256, 256, 0, stream>>>(w2, B2p, 256, 256, 768, 256*768);
  pack_w_f16<<<(224*768 + 255)/256, 256, 0, stream>>>(w3, B3p, 222, 256, 768, 224*768);

  // conv1: im2col(x) -> A1, then WMMA GEMM (M=256,N=256,Kc=240->256), ReLU
  pack_a_f16< 80><<<(256*256 + 255)/256, 256, 0, stream>>>(x,  A1p, 256, 256*256);
  gemm_wmma_f16<true,  false><<<32, 128, 0, stream>>>(A1p, B1p, b1, nullptr, h1, 256, 16, 256, 256);

  // conv2: im2col(h1) -> A2, GEMM (M=256,N=256,Kc=768), ReLU
  pack_a_f16<256><<<(256*768 + 255)/256, 256, 0, stream>>>(h1, A2p, 768, 256*768);
  gemm_wmma_f16<true,  false><<<32, 128, 0, stream>>>(A2p, B2p, b2, nullptr, h2, 768, 16, 256, 256);

  // conv3: im2col(h2) -> A3, GEMM (M=256,N=222->224,Kc=768), divide quef_norm
  pack_a_f16<256><<<(256*768 + 255)/256, 256, 0, stream>>>(h2, A3p, 768, 256*768);
  gemm_wmma_f16<false, true ><<<28, 128, 0, stream>>>(A3p, B3p, b3, quef,    cc, 768, 14, 222, 224);

  // FFT -> spectral transform -> IFFT (real part) per frame
  cepstrum_fft<<<256, 256, 0, stream>>>(cc, imp);

  // Time-varying FIR per frame + window
  ltv_filter<<<256, 256, 0, stream>>>(imp, z, win, outw);

  // Overlap-add
  overlap_add<<<ZLEN/256, 256, 0, stream>>>(outw, sbuf);

  // 128-tap post filter -> output
  post_filter<<<ZLEN/256, 256, 0, stream>>>(sbuf, pfw, pfb, yout);
}
